// AdvancedPhaseCoherence_34187939676375
// MI455X (gfx1250) — compile-verified, matrix operand order checked by reading, not run on hardware
//
#include <hip/hip_runtime.h>
#include <hip/hip_bf16.h>
#include <math.h>

// ---------------------------------------------------------------------------
// Problem constants (from the reference)
// ---------------------------------------------------------------------------
constexpr int Bq = 4;
constexpr int Sq = 2048;
constexpr int Dq = 512;
constexpr int Vq = 4096;
constexpr int SEMW = 8;          // semantic half-window
constexpr int CTXW = 8;          // context half-window
constexpr int NOFF = 2 * SEMW + 1;  // 17

typedef __attribute__((ext_vector_type(16))) __bf16 v16bf;
typedef __attribute__((ext_vector_type(8)))  float  v8f;

// ---------------------------------------------------------------------------
// Helpers
// ---------------------------------------------------------------------------
__device__ __forceinline__ float block_reduce(float v, float* sh) {
    int t = threadIdx.x;
    sh[t] = v;
    __syncthreads();
    for (int s = blockDim.x >> 1; s > 0; s >>= 1) {
        if (t < s) sh[t] += sh[t + s];
        __syncthreads();
    }
    float r = sh[0];
    __syncthreads();
    return r;
}

__device__ __forceinline__ float gelu_exact(float x) {
    return 0.5f * x * (1.0f + erff(x * 0.70710678118654752440f));
}

// ---------------------------------------------------------------------------
// Stage 1: phases = tanh(embeddings)
// ---------------------------------------------------------------------------
__global__ __launch_bounds__(256) void k_phase(const float* __restrict__ emb,
                                               float* __restrict__ ph) {
    size_t idx = (size_t)blockIdx.x * blockDim.x + threadIdx.x;
    ph[idx] = tanhf(emb[idx]);
}

// ---------------------------------------------------------------------------
// Stage 2: reciprocal row norms of token_similarity (V rows of V)
// ---------------------------------------------------------------------------
__global__ __launch_bounds__(256) void k_rownorm(const float* __restrict__ sim,
                                                 float* __restrict__ invn) {
    __shared__ float sdata[256];
    int v = blockIdx.x;
    const float* row = sim + (size_t)v * Vq;
    float s = 0.f;
    for (int i = threadIdx.x; i < Vq; i += 256) {
        float t = row[i];
        s += t * t;
    }
    float tot = block_reduce(s, sdata);
    if (threadIdx.x == 0) invn[v] = 1.f / fmaxf(sqrtf(tot), 1e-12f);
}

// ---------------------------------------------------------------------------
// Stage 3: banded token similarity.
// Only the +-8 diagonal of the S x S gram matrix is ever consumed, so compute
// just the band: 17 dots of length V per (b,s). token_similarity (67 MB) lives
// in the 192 MB L2, so neighbor-row streams are L2 hits.
// ---------------------------------------------------------------------------
__global__ __launch_bounds__(128) void k_band(const float* __restrict__ sim,
                                              const int* __restrict__ ids,
                                              const float* __restrict__ invn,
                                              float* __restrict__ band) {
    __shared__ float rowA[Vq];     // 16 KB anchor row
    __shared__ float red[128];
    int bs = blockIdx.x;
    int s = bs % Sq, b = bs / Sq;
    int idA = ids[bs];
    const float* rA = sim + (size_t)idA * Vq;
    for (int i = threadIdx.x; i < Vq; i += 128) rowA[i] = rA[i];
    __syncthreads();
    float iA = invn[idA];
    for (int o = 0; o < NOFF; ++o) {
        int sc = s + o - SEMW;
        sc = sc < 0 ? 0 : (sc >= Sq ? Sq - 1 : sc);
        int idB = ids[b * Sq + sc];
        const float* rB = sim + (size_t)idB * Vq;
        __builtin_prefetch(rB + threadIdx.x, 0, 1);   // global_prefetch_b8
        float acc = 0.f;
        for (int i = threadIdx.x; i < Vq; i += 128) acc += rowA[i] * rB[i];
        red[threadIdx.x] = acc;
        __syncthreads();
        for (int st = 64; st > 0; st >>= 1) {
            if (threadIdx.x < st) red[threadIdx.x] += red[threadIdx.x + st];
            __syncthreads();
        }
        if (threadIdx.x == 0) band[(size_t)bs * NOFF + o] = red[0] * iA * invn[idB];
        __syncthreads();
    }
}

// ---------------------------------------------------------------------------
// Stage 4: semantic coherence (shared across batch) + write sem_coh column.
// ---------------------------------------------------------------------------
__global__ __launch_bounds__(256) void k_sem(const float* __restrict__ ph,
                                             const float* __restrict__ band,
                                             const float* __restrict__ semW,
                                             const float* __restrict__ semb,
                                             float* __restrict__ allc) {
    __shared__ float sdata[256];
    int s = blockIdx.x;
    int nvalid = 0;
    for (int o = 0; o < NOFF; ++o) {
        int t = s + o - SEMW;
        if (t >= 0 && t < Sq) nvalid++;
    }
    float acc = 0.f;
    for (int b = 0; b < Bq; ++b) {
        const float* pb = ph + (size_t)b * Sq * Dq;
        const float* p0 = pb + (size_t)s * Dq;
        for (int o = 0; o < NOFF; ++o) {
            int t = s + o - SEMW;
            if (t < 0 || t >= Sq) continue;          // mask==0 entries skipped
            float w = band[(size_t)(b * Sq + s) * NOFF + o];
            const float* p1 = pb + (size_t)t * Dq;
            for (int d = threadIdx.x; d < Dq; d += 256)
                acc += cosf((p0[d] - p1[d]) * w);
        }
    }
    float tot = block_reduce(acc, sdata);
    float semv = tot / ((float)nvalid * (float)(Bq * Dq));
    for (int b = 0; b < Bq; ++b) {
        float* dst = allc + (size_t)(b * Sq + s) * (4 * Dq) + Dq;
        for (int d = threadIdx.x; d < Dq; d += 256)
            dst[d] = semv * semW[d] + semb[d];
    }
}

// ---------------------------------------------------------------------------
// Stage 5: multi-scale coherence + tiny K=5 projection, fused into allc col 0.
// ---------------------------------------------------------------------------
__global__ __launch_bounds__(256) void k_msc(const float* __restrict__ ph,
                                             const float* __restrict__ scaleW,
                                             const float* __restrict__ scaleb,
                                             float* __restrict__ allc) {
    __shared__ float sdata[256];
    __shared__ float cohs[5];
    int bs = blockIdx.x;
    int s = bs % Sq, b = bs / Sq;
    const float* pb = ph + (size_t)b * Sq * Dq;
    const int scales[5] = {1, 2, 4, 8, 16};
    for (int si = 0; si < 5; ++si) {
        int sl = scales[si];
        int L = Sq - sl + 1;
        float part = 0.f;
        if (s < L) {
            for (int dd = 0; dd < 2; ++dd) {
                int d = threadIdx.x + dd * 256;
                float wsum = 0.f;
                for (int k = 0; k < sl; ++k) wsum += pb[(size_t)(s + k) * Dq + d];
                float wm = wsum / (float)sl;
                for (int k = 0; k < sl; ++k)
                    part += cosf(pb[(size_t)(s + k) * Dq + d] - wm);
            }
        }
        float tot = block_reduce(part, sdata);
        if (threadIdx.x == 0) cohs[si] = (s < L) ? tot / (float)(sl * Dq) : 0.f;
        __syncthreads();
    }
    for (int dd = 0; dd < 2; ++dd) {
        int d = threadIdx.x + dd * 256;
        float v = scaleb[d];
        for (int j = 0; j < 5; ++j) v += cohs[j] * scaleW[j * Dq + d];
        allc[(size_t)bs * (4 * Dq) + d] = v;
    }
}

// ---------------------------------------------------------------------------
// Stage 6: context-window cosine deviation (GEMM input)
// ---------------------------------------------------------------------------
__global__ __launch_bounds__(256) void k_ctx(const float* __restrict__ emb,
                                             float* __restrict__ ctxin) {
    size_t idx = (size_t)blockIdx.x * blockDim.x + threadIdx.x;
    int d = (int)(idx % Dq);
    int s = (int)((idx / Dq) % Sq);
    int b = (int)(idx / ((size_t)Dq * Sq));
    int s0 = (s - CTXW < 0) ? 0 : s - CTXW;
    int e0 = (s + CTXW + 1 > Sq) ? Sq : s + CTXW + 1;
    const float* eb = emb + (size_t)b * Sq * Dq + d;
    float sum = 0.f;
    for (int t = s0; t < e0; ++t) sum += eb[(size_t)t * Dq];
    float wm = sum / (float)(e0 - s0);
    ctxin[idx] = cosf(emb[idx] - wm);
}

// ---------------------------------------------------------------------------
// Stage 7: local coherence -> allc col 3
// ---------------------------------------------------------------------------
__global__ __launch_bounds__(256) void k_loc(const float* __restrict__ emb,
                                             float* __restrict__ allc) {
    size_t idx = (size_t)blockIdx.x * blockDim.x + threadIdx.x;
    int d = (int)(idx % Dq);
    int s = (int)((idx / Dq) % Sq);
    int b = (int)(idx / ((size_t)Dq * Sq));
    float v = 0.f;
    if (s < Sq - 1) v = cosf(emb[idx + Dq] - emb[idx]);
    allc[(size_t)(b * Sq + s) * (4 * Dq) + 3 * Dq + d] = v;
}

// ---------------------------------------------------------------------------
// Weight packing: f32 (K x 512) row-major -> bf16 in the exact WMMA B-operand
// per-lane layout: [nTile][kStep][lane][16 elems], so the GEMM's B fragment is
// two contiguous b128 loads per lane.
// 16-bit operand K-packing (ISA 7.12.2): lanes 0-15 hold K in {0..7,16..23},
// lanes 16-31 hold K in {8..15,24..31} of each 32-wide K step.
// ---------------------------------------------------------------------------
__global__ __launch_bounds__(256) void k_pack_w(const float* __restrict__ W,
                                                unsigned short* __restrict__ out,
                                                int K) {
    int idx = blockIdx.x * blockDim.x + threadIdx.x;   // K*512 total
    if (idx >= K * Dq) return;
    int j    = idx & 15;
    int lane = (idx >> 4) & 31;
    int blk  = idx >> 9;
    int ksteps = K >> 5;
    int ks = blk % ksteps;
    int nT = blk / ksteps;
    int n  = nT * 16 + (lane & 15);
    int kb = (lane >> 4) * 8;
    int kk = ks * 32 + ((j < 8) ? (kb + j) : (16 + kb + (j - 8)));
    __bf16 h = (__bf16)W[(size_t)kk * Dq + n];
    out[idx] = *(const unsigned short*)&h;
}

// ---------------------------------------------------------------------------
// WMMA GEMM: C[M,512] = A[M,K] * B[K,512] (+ bias), f32 in / bf16 MACs /
// f32 accumulate on v_wmma_f32_16x16x32_bf16.
//
// Tiling: each wave owns a 16 (M) x 64 (N) strip = 4 accumulators. One A
// fragment per k-step is shared by 4 WMMAs against 4 contiguous pre-packed
// B fragments -> 4x less A load/convert work per FLOP than 16x16-per-wave,
// and a 4x higher wmma-to-vmem ratio. 8 waves / block, one strip per wave.
// ---------------------------------------------------------------------------
__global__ __launch_bounds__(256) void k_gemm_bf16(const float* __restrict__ A,
                                                   const unsigned short* __restrict__ Bp,
                                                   const float* __restrict__ bias,
                                                   float* __restrict__ C,
                                                   int M, int K, int lda, int ldc) {
    const int lane = threadIdx.x & 31;
    const int wid  = blockIdx.x * 8 + (threadIdx.x >> 5);
    const int stripsN = Dq / 64;                // 8 strips of 4 N-tiles
    const int strip  = wid % stripsN;
    const int tileM  = wid / stripsN;
    if (tileM * 16 >= M) return;

    const int r  = lane & 15;
    const int hi = lane >> 4;
    const int kb = hi * 8;
    const int ksteps = K >> 5;
    const float* arow = A + (size_t)(tileM * 16 + r) * lda;
    const v16bf* bp = (const v16bf*)Bp;
    // packed B fragment index for (tileN, ks): (tileN*ksteps + ks)*32 + lane
    const int tileN0 = strip * 4;

    v8f acc0 = {0.f, 0.f, 0.f, 0.f, 0.f, 0.f, 0.f, 0.f};
    v8f acc1 = acc0, acc2 = acc0, acc3 = acc0;

    for (int ks = 0; ks < ksteps; ++ks) {
        const float* ap = arow + ks * 32 + kb;
        v16bf av;
#pragma unroll
        for (int j = 0; j < 8; ++j) av[j] = (__bf16)ap[j];
#pragma unroll
        for (int j = 0; j < 8; ++j) av[8 + j] = (__bf16)ap[16 + j];
        const size_t bofs = (size_t)ks * 32 + lane;
        v16bf bv0 = bp[(size_t)(tileN0 + 0) * ksteps * 32 + bofs];
        v16bf bv1 = bp[(size_t)(tileN0 + 1) * ksteps * 32 + bofs];
        v16bf bv2 = bp[(size_t)(tileN0 + 2) * ksteps * 32 + bofs];
        v16bf bv3 = bp[(size_t)(tileN0 + 3) * ksteps * 32 + bofs];
        acc0 = __builtin_amdgcn_wmma_f32_16x16x32_bf16(false, av, false, bv0,
                                                       (short)0, acc0, false, false);
        acc1 = __builtin_amdgcn_wmma_f32_16x16x32_bf16(false, av, false, bv1,
                                                       (short)0, acc1, false, false);
        acc2 = __builtin_amdgcn_wmma_f32_16x16x32_bf16(false, av, false, bv2,
                                                       (short)0, acc2, false, false);
        acc3 = __builtin_amdgcn_wmma_f32_16x16x32_bf16(false, av, false, bv3,
                                                       (short)0, acc3, false, false);
    }

    // Epilogue: C/D layout -> VGPR i holds M = tileM*16 + i + 8*hi, N = lane&15
    v8f accs[4] = {acc0, acc1, acc2, acc3};
#pragma unroll
    for (int nn = 0; nn < 4; ++nn) {
        const int n = (tileN0 + nn) * 16 + r;
        const float bb = bias ? bias[n] : 0.f;
#pragma unroll
        for (int i = 0; i < 8; ++i) {
            int m = tileM * 16 + i + 8 * hi;
            C[(size_t)m * ldc + n] = accs[nn][i] + bb;
        }
    }
}

// ---------------------------------------------------------------------------
// LayerNorm over D=512 (one row per block, 2 elems/thread), optional GELU.
// ---------------------------------------------------------------------------
__global__ __launch_bounds__(256) void k_layernorm(const float* __restrict__ in,
                                                   const float* __restrict__ g,
                                                   const float* __restrict__ bta,
                                                   float* __restrict__ out,
                                                   int do_gelu) {
    __shared__ float sdata[256];
    const float* x = in + (size_t)blockIdx.x * Dq;
    float v0 = x[threadIdx.x];
    float v1 = x[threadIdx.x + 256];
    float m = block_reduce(v0 + v1, sdata) * (1.f / Dq);
    float d0 = v0 - m, d1 = v1 - m;
    float var = block_reduce(d0 * d0 + d1 * d1, sdata) * (1.f / Dq);
    float inv = rsqrtf(var + 1e-5f);
    float* y = out + (size_t)blockIdx.x * Dq;
    float r0 = d0 * inv * g[threadIdx.x] + bta[threadIdx.x];
    float r1 = d1 * inv * g[threadIdx.x + 256] + bta[threadIdx.x + 256];
    if (do_gelu) { r0 = gelu_exact(r0); r1 = gelu_exact(r1); }
    y[threadIdx.x] = r0;
    y[threadIdx.x + 256] = r1;
}

// ---------------------------------------------------------------------------
// Residual: x += 0.1 * gelu(h)
// ---------------------------------------------------------------------------
__global__ __launch_bounds__(256) void k_enh_post(float* __restrict__ x,
                                                  const float* __restrict__ h) {
    size_t idx = (size_t)blockIdx.x * blockDim.x + threadIdx.x;
    x[idx] += 0.1f * gelu_exact(h[idx]);
}

// ---------------------------------------------------------------------------
// Launch
// ---------------------------------------------------------------------------
extern "C" void kernel_launch(void* const* d_in, const int* in_sizes, int n_in,
                              void* d_out, int out_size, void* d_ws, size_t ws_size,
                              hipStream_t stream) {
    const float* emb     = (const float*)d_in[0];
    const int*   ids     = (const int*)  d_in[1];
    const float* tsim    = (const float*)d_in[2];
    const float* scale_W = (const float*)d_in[3];
    const float* scale_b = (const float*)d_in[4];
    const float* sem_W   = (const float*)d_in[5];
    const float* sem_b   = (const float*)d_in[6];
    const float* ctx_W   = (const float*)d_in[7];
    const float* ctx_b   = (const float*)d_in[8];
    const float* fus_W   = (const float*)d_in[9];
    const float* fus_b   = (const float*)d_in[10];
    const float* enh_W   = (const float*)d_in[11];
    const float* enh_b   = (const float*)d_in[12];
    const float* ln1_g   = (const float*)d_in[13];
    const float* ln1_b   = (const float*)d_in[14];
    const float* ln2_g   = (const float*)d_in[15];
    const float* ln2_b   = (const float*)d_in[16];
    const float* ln3_g   = (const float*)d_in[17];
    const float* ln3_b   = (const float*)d_in[18];
    float* out = (float*)d_out;

    const size_t BSD = (size_t)Bq * Sq * Dq;           // 4,194,304

    // Workspace carve-out (256B aligned slabs)
    char* w = (char*)d_ws;
    auto alloc = [&](size_t bytes) {
        char* p = w;
        w += (bytes + 255) & ~(size_t)255;
        return p;
    };
    float* ph    = (float*)alloc(BSD * 4);
    float* allc  = (float*)alloc(BSD * 4 * 4);         // (B*S, 4D)
    float* ctxin = (float*)alloc(BSD * 4);
    float* ybuf  = (float*)alloc(BSD * 4);
    float* xbuf  = (float*)alloc(BSD * 4);
    float* lnbuf = (float*)alloc(BSD * 4);
    float* hbuf  = (float*)alloc(BSD * 4);
    float* band  = (float*)alloc((size_t)Bq * Sq * NOFF * 4);
    float* invn  = (float*)alloc((size_t)Vq * 4);
    unsigned short* pw_ctx = (unsigned short*)alloc((size_t)Dq * Dq * 2);
    unsigned short* pw_fus = (unsigned short*)alloc((size_t)4 * Dq * Dq * 2);
    unsigned short* pw_enh = (unsigned short*)alloc((size_t)3 * Dq * Dq * 2);

    const int M = Bq * Sq;                              // 8192 rows
    const int elemBlocks = (int)(BSD / 256);            // 16384

    // Elementwise / coherence stages
    k_phase  <<<elemBlocks, 256, 0, stream>>>(emb, ph);
    k_rownorm<<<Vq,         256, 0, stream>>>(tsim, invn);
    k_band   <<<Bq * Sq,    128, 0, stream>>>(tsim, ids, invn, band);
    k_sem    <<<Sq,         256, 0, stream>>>(ph, band, sem_W, sem_b, allc);
    k_msc    <<<Bq * Sq,    256, 0, stream>>>(ph, scale_W, scale_b, allc);
    k_ctx    <<<elemBlocks, 256, 0, stream>>>(emb, ctxin);
    k_loc    <<<elemBlocks, 256, 0, stream>>>(emb, allc);

    // Pack weights into WMMA B-operand layout (bf16)
    k_pack_w<<<(Dq * Dq) / 256,     256, 0, stream>>>(ctx_W, pw_ctx, Dq);
    k_pack_w<<<(4 * Dq * Dq) / 256, 256, 0, stream>>>(fus_W, pw_fus, 4 * Dq);
    for (int i = 0; i < 3; ++i)
        k_pack_w<<<(Dq * Dq) / 256, 256, 0, stream>>>(
            enh_W + (size_t)i * Dq * Dq, pw_enh + (size_t)i * Dq * Dq, Dq);

    // GEMM grid: one 16x64 strip per wave, 8 waves per block
    const int strips  = (M / 16) * (Dq / 64);           // 4096 wave-strips
    const int gblocks = strips / 8;                     // 512 blocks

    // ctx GEMM -> allc column 2 (ldc = 4D)
    k_gemm_bf16<<<gblocks, 256, 0, stream>>>(ctxin, pw_ctx, ctx_b,
                                             allc + 2 * Dq, M, Dq, Dq, 4 * Dq);
    // fusion GEMM: allc (M x 4D) @ fus_W -> ybuf
    k_gemm_bf16<<<gblocks, 256, 0, stream>>>(allc, pw_fus, fus_b,
                                             ybuf, M, 4 * Dq, 4 * Dq, Dq);
    // x = gelu(LN1(ybuf))
    k_layernorm<<<M, 256, 0, stream>>>(ybuf, ln1_g, ln1_b, xbuf, 1);

    // 3 enhancement blocks: x += 0.1 * gelu(LN2(x) @ W_i + b_i)
    for (int i = 0; i < 3; ++i) {
        k_layernorm<<<M, 256, 0, stream>>>(xbuf, ln2_g, ln2_b, lnbuf, 0);
        k_gemm_bf16<<<gblocks, 256, 0, stream>>>(lnbuf,
                                                 pw_enh + (size_t)i * Dq * Dq,
                                                 enh_b + (size_t)i * Dq,
                                                 hbuf, M, Dq, Dq, Dq);
        k_enh_post<<<elemBlocks, 256, 0, stream>>>(xbuf, hbuf);
    }

    // Final LayerNorm -> output
    k_layernorm<<<M, 256, 0, stream>>>(xbuf, ln3_g, ln3_b, out, 0);
}
